// StatefulQwen2Decoder_60095182405741
// MI455X (gfx1250) — matrix operand order repaired and checked by reading, not compiled, not run on hardware
//
#include <hip/hip_runtime.h>

// ---------------------------------------------------------------------------
// Qwen2 4-layer decoder forward for MI455X (gfx1250), wave32 + WMMA f16.
// Data movement: GLOBAL_LOAD_ASYNC_TO_LDS_B128 (ASYNCcnt) for all GEMM /
// attention tile staging; weights pre-transposed at f32->f16 conversion so
// every LDS stage is a natural row-major B128 copy. GEMM: 128x64 block tile,
// K-step 64, double-buffered (8 WMMAs per barrier window).
// ---------------------------------------------------------------------------

typedef __attribute__((ext_vector_type(16))) _Float16 v16h;
typedef __attribute__((ext_vector_type(8)))  float    v8f;

union Frag16 { uint4 u4[2]; v16h v; };

constexpr int   Lc    = 4;
constexpr int   Hc    = 1536;
constexpr int   NQc   = 12;
constexpr int   NKVc  = 2;
constexpr int   Dc    = 128;
constexpr int   Ic    = 8960;
constexpr int   Sc    = 2048;
constexpr float EPSc  = 1e-6f;
constexpr float SCALEc = 0.08838834764831845f; // 1/sqrt(128)

__device__ __forceinline__ v8f wmma_f16(v16h a, v16h b, v8f c) {
  return __builtin_amdgcn_wmma_f32_16x16x32_f16(
      /*neg_a=*/false, a, /*neg_b=*/false, b,
      /*c_mod=*/(short)0, c, /*reuse_a=*/false, /*reuse_b=*/false);
}

// Async global->LDS copy of 16 bytes per lane (ASYNCcnt-tracked).
__device__ __forceinline__ void async_b128(unsigned lds, const void* g) {
  asm volatile("global_load_async_to_lds_b128 %0, %1, off"
               :: "v"(lds), "v"(g) : "memory");
}
__device__ __forceinline__ void wait_async() {
  asm volatile("s_wait_asynccnt 0" ::: "memory");
}
__device__ __forceinline__ unsigned lds_addr(const void* p) {
  return (unsigned)(unsigned long long)p;  // LDS aperture: offset = addr[31:0]
}

// ---------------------------------------------------------------------------
// Transposing f32 -> f16 weight conversion: in W[K,N] f32 -> out Wt[N,K] f16.
// Grid (N/32, K/32), block (32,8).
// ---------------------------------------------------------------------------
__global__ void convT_kernel(const float* __restrict__ in,
                             _Float16* __restrict__ out, int K, int N) {
  __shared__ float t[32][33];
  int n0 = blockIdx.x * 32, k0 = blockIdx.y * 32;
  for (int i = threadIdx.y; i < 32; i += 8)
    t[i][threadIdx.x] = in[(long)(k0 + i) * N + n0 + threadIdx.x];
  __syncthreads();
  for (int i = threadIdx.y; i < 32; i += 8)
    out[(long)(n0 + i) * K + k0 + threadIdx.x] = (_Float16)t[threadIdx.x][i];
}

// ---------------------------------------------------------------------------
// RMSNorm: one block per token row, H=1536
// ---------------------------------------------------------------------------
template <typename OT>
__global__ void rmsnorm_kernel(const float* __restrict__ x,
                               const float* __restrict__ w,
                               OT* __restrict__ out) {
  int row = blockIdx.x;
  const float* xr = x + (long)row * Hc;
  __shared__ float red[256];
  float ss = 0.f;
  for (int j = threadIdx.x; j < Hc; j += 256) { float v = xr[j]; ss += v * v; }
  red[threadIdx.x] = ss;
  __syncthreads();
  for (int s = 128; s > 0; s >>= 1) {
    if (threadIdx.x < s) red[threadIdx.x] += red[threadIdx.x + s];
    __syncthreads();
  }
  float inv = rsqrtf(red[0] / (float)Hc + EPSc);
  OT* orow = out + (long)row * Hc;
  for (int j = threadIdx.x; j < Hc; j += 256)
    orow[j] = (OT)(xr[j] * inv * w[j]);
}

// ---------------------------------------------------------------------------
// RoPE: in [S, nh*D] fp32 -> out [nh][S][D] f16 (head-major)
// ---------------------------------------------------------------------------
__global__ void rope_kernel(const float* __restrict__ in,
                            const float* __restrict__ cosp,
                            const float* __restrict__ sinp,
                            _Float16* __restrict__ out, int nh) {
  int idx = blockIdx.x * blockDim.x + threadIdx.x;
  int total = Sc * nh * 64;
  if (idx >= total) return;
  int d = idx & 63;
  int h = (idx >> 6) % nh;
  int s = idx / (64 * nh);
  long ib = (long)s * (nh * Dc) + h * Dc;
  float x1 = in[ib + d], x2 = in[ib + d + 64];
  float c1 = cosp[s * Dc + d],      s1 = sinp[s * Dc + d];
  float c2 = cosp[s * Dc + d + 64], s2 = sinp[s * Dc + d + 64];
  long ob = ((long)h * Sc + s) * Dc;
  out[ob + d]      = (_Float16)(x1 * c1 - x2 * s1);
  out[ob + d + 64] = (_Float16)(x2 * c2 + x1 * s2);
}

// V: [S, NKV*D] fp32 -> [NKV][D][S] f16 (d-major so V tiles stage row-major)
__global__ void v_pack_kernel(const float* __restrict__ in,
                              _Float16* __restrict__ out) {
  int idx = blockIdx.x * blockDim.x + threadIdx.x;
  if (idx >= NKVc * Dc * Sc) return;
  int s = idx & (Sc - 1);
  int d = (idx >> 11) & (Dc - 1);
  int h = idx >> 18;
  out[((long)h * Dc + d) * Sc + s] = (_Float16)in[(long)s * NKVc * Dc + h * Dc + d];
}

// SwiGLU: act = silu(g) * u, f16 out
__global__ void swiglu_kernel(const float* __restrict__ g,
                              const float* __restrict__ u,
                              _Float16* __restrict__ out, long n) {
  long i  = (long)blockIdx.x * blockDim.x + threadIdx.x;
  long st = (long)gridDim.x * blockDim.x;
  for (; i < n; i += st) {
    float gv = g[i];
    float sv = gv / (1.f + __expf(-gv));
    out[i] = (_Float16)(sv * u[i]);
  }
}

// ---------------------------------------------------------------------------
// WMMA GEMM: C[M,N] f32 = A[M,K] f16 * Bt[N,K]^T f16 (+bias[N]) (+res[M,N]).
// Block 256 thr (8 waves), tile 128x64, K-step 64, double-buffered LDS with
// async global->LDS staging. M%128==0, N%64==0, K%64==0.
// ---------------------------------------------------------------------------
template <bool BIAS, bool RES>
__global__ __launch_bounds__(256)
void gemm_f16_kernel(const _Float16* __restrict__ A,
                     const _Float16* __restrict__ Bt,
                     float* __restrict__ C,
                     const float* __restrict__ bias,
                     const float* __restrict__ res,
                     int M, int N, int K) {
  __shared__ alignas(16) _Float16 As[2][128][72];  // [m][k], pitch 144B
  __shared__ alignas(16) _Float16 Bs[2][64][72];   // [n][k], pitch 144B
  int tid  = threadIdx.x;
  int lane = tid & 31, w = tid >> 5;
  int g = lane >> 4, ln = lane & 15;
  int bm = blockIdx.y * 128, bn = blockIdx.x * 64;
  int wm = (w & 3) * 32, wn = (w >> 2) * 32;

  v8f acc[2][2] = {};

  auto issue = [&](int k0, int b) {
    // A tile 128x64 halves = 1024 B128 chunks
#pragma unroll
    for (int i = 0; i < 4; i++) {
      int idx = tid + i * 256;
      int row = idx >> 3, cc = (idx & 7) * 8;
      async_b128(lds_addr(&As[b][row][cc]),
                 A + (long)(bm + row) * K + k0 + cc);
    }
    // Bt tile 64x64 halves = 512 B128 chunks
#pragma unroll
    for (int i = 0; i < 2; i++) {
      int idx = tid + i * 256;
      int n = idx >> 3, cc = (idx & 7) * 8;
      async_b128(lds_addr(&Bs[b][n][cc]),
                 Bt + (long)(bn + n) * K + k0 + cc);
    }
  };

  int niter = K >> 6;
  issue(0, 0);
  for (int it = 0; it < niter; it++) {
    wait_async();
    __syncthreads();
    if (it + 1 < niter) issue((it + 1) << 6, (it + 1) & 1);
    int b = it & 1;

#pragma unroll
    for (int kk = 0; kk < 2; kk++) {  // two 32-wide K chunks per stage
      Frag16 af[2], bf[2];
#pragma unroll
      for (int mt = 0; mt < 2; mt++) {
        int row = wm + mt * 16 + ln;
        // A lane layout: K = e%8 + (e/8)*16 + g*8
        af[mt].u4[0] = *(const uint4*)(&As[b][row][kk * 32 + g * 8]);
        af[mt].u4[1] = *(const uint4*)(&As[b][row][kk * 32 + g * 8 + 16]);
      }
#pragma unroll
      for (int nt = 0; nt < 2; nt++) {
        int col = wn + nt * 16 + ln;
        // B lane layout: N = lane%16, K = e%8 + (e/8)*16 + g*8
        bf[nt].u4[0] = *(const uint4*)(&Bs[b][col][kk * 32 + g * 8]);
        bf[nt].u4[1] = *(const uint4*)(&Bs[b][col][kk * 32 + g * 8 + 16]);
      }
#pragma unroll
      for (int mt = 0; mt < 2; mt++)
#pragma unroll
        for (int nt = 0; nt < 2; nt++)
          acc[mt][nt] = wmma_f16(af[mt].v, bf[nt].v, acc[mt][nt]);
    }
    __syncthreads();
  }

  // Epilogue. C layout: VGPR r -> M = r + g*8, N = lane%16.
#pragma unroll
  for (int mt = 0; mt < 2; mt++)
#pragma unroll
    for (int nt = 0; nt < 2; nt++) {
      int n = bn + wn + nt * 16 + ln;
#pragma unroll
      for (int r = 0; r < 8; r++) {
        int m = bm + wm + mt * 16 + r + g * 8;
        float v = acc[mt][nt][r];
        if (BIAS) v += bias[n];
        if (RES)  v += res[(long)m * N + n];
        C[(long)m * N + n] = v;
      }
    }
}

// ---------------------------------------------------------------------------
// Flash attention (causal, GQA). Q:[NQ][S][D] f16, K:[NKV][S][D] f16,
// V:[NKV][D][S] f16 (d-major), Out:[S][NQ*D] f16.
// Grid (S/64, NQ), block 128 (4 waves, one 16-row Q strip per wave).
// K/V tiles staged via async global->LDS; QK^T and P*V via WMMA.
// ---------------------------------------------------------------------------
__global__ __launch_bounds__(128)
void flash_attn_kernel(const _Float16* __restrict__ Q,
                       const _Float16* __restrict__ Kh,
                       const _Float16* __restrict__ Vh,
                       _Float16* __restrict__ Out) {
  __shared__ alignas(16) _Float16 Ks[32][136];    // [key][d]   pitch 272B
  __shared__ alignas(16) _Float16 Vs[128][40];    // [d][key]   pitch 80B
  __shared__ alignas(16) _Float16 Ps[4][16][40];  // per-wave P bounce [m][key]

  int h   = blockIdx.y;
  int kvh = h / (NQc / NKVc);
  int qt  = blockIdx.x;
  int qbase = qt * 64;
  int tid  = threadIdx.x;
  int lane = tid & 31, w = tid >> 5;
  int g = lane >> 4, ln = lane & 15;
  int qrow0 = qbase + w * 16;

  const _Float16* Qh = Q  + (long)h   * Sc * Dc;
  const _Float16* Kp = Kh + (long)kvh * Sc * Dc;
  const _Float16* Vt = Vh + (long)kvh * Dc * Sc;  // [D][S]

  // Resident Q fragments: 4 chunks of 16x32 over D=128
  Frag16 qf[4];
#pragma unroll
  for (int c = 0; c < 4; c++) {
    const _Float16* qr = Qh + (long)(qrow0 + ln) * Dc + c * 32;
    qf[c].u4[0] = *(const uint4*)(qr + g * 8);
    qf[c].u4[1] = *(const uint4*)(qr + g * 8 + 16);
  }

  v8f o[8] = {};
  float mrun[8], lrun[8];
#pragma unroll
  for (int r = 0; r < 8; r++) { mrun[r] = -1e30f; lrun[r] = 0.f; }

  int ntiles = qt * 2 + 2;  // causal: keys up to qbase+63
  for (int j = 0; j < ntiles; j++) {
    int kb = j * 32;
    __syncthreads();
    // Stage K tile [32 keys][128 d]: 512 chunks, async
#pragma unroll
    for (int i = 0; i < 4; i++) {
      int idx = tid + i * 128;
      int kr = idx >> 4, cc = (idx & 15) * 8;
      async_b128(lds_addr(&Ks[kr][cc]), Kp + (long)(kb + kr) * Dc + cc);
    }
    // Stage V tile [128 d][32 keys]: 512 chunks, async (V is d-major global)
#pragma unroll
    for (int i = 0; i < 4; i++) {
      int idx = tid + i * 128;
      int dr = idx >> 2, cc = (idx & 3) * 8;
      async_b128(lds_addr(&Vs[dr][cc]), Vt + (long)dr * Sc + kb + cc);
    }
    wait_async();
    __syncthreads();

    // S = Q * K^T : two 16x16 score tiles over 32 keys
    v8f sc[2] = {};
#pragma unroll
    for (int c = 0; c < 4; c++) {
#pragma unroll
      for (int nt = 0; nt < 2; nt++) {
        Frag16 bf;
        int key = nt * 16 + ln;
        bf.u4[0] = *(const uint4*)(&Ks[key][c * 32 + g * 8]);
        bf.u4[1] = *(const uint4*)(&Ks[key][c * 32 + g * 8 + 16]);
        sc[nt] = wmma_f16(qf[c].v, bf.v, sc[nt]);
      }
    }

    // Scale + causal mask
    float p[2][8];
#pragma unroll
    for (int nt = 0; nt < 2; nt++) {
      int kg = kb + nt * 16 + ln;
#pragma unroll
      for (int r = 0; r < 8; r++) {
        int qg = qrow0 + r + g * 8;
        p[nt][r] = sc[nt][r] * SCALEc + (kg <= qg ? 0.f : -1e9f);
      }
    }

    // Online softmax. Row M=r+g*8 lives in 16 lanes of one half-wave.
#pragma unroll
    for (int r = 0; r < 8; r++) {
      float tm = fmaxf(p[0][r], p[1][r]);
      tm = fmaxf(tm, __shfl_xor(tm, 1, 32));
      tm = fmaxf(tm, __shfl_xor(tm, 2, 32));
      tm = fmaxf(tm, __shfl_xor(tm, 4, 32));
      tm = fmaxf(tm, __shfl_xor(tm, 8, 32));
      float mnew = fmaxf(mrun[r], tm);
      float corr = __expf(mrun[r] - mnew);
      mrun[r] = mnew;
      p[0][r] = __expf(p[0][r] - mnew);
      p[1][r] = __expf(p[1][r] - mnew);
      float rs = p[0][r] + p[1][r];
      rs += __shfl_xor(rs, 1, 32);
      rs += __shfl_xor(rs, 2, 32);
      rs += __shfl_xor(rs, 4, 32);
      rs += __shfl_xor(rs, 8, 32);
      lrun[r] = lrun[r] * corr + rs;
#pragma unroll
      for (int nd = 0; nd < 8; nd++) o[nd][r] *= corr;
    }

    // C-layout -> A-layout transpose of P via per-wave LDS bounce
#pragma unroll
    for (int nt = 0; nt < 2; nt++)
#pragma unroll
      for (int r = 0; r < 8; r++)
        Ps[w][r + g * 8][nt * 16 + ln] = (_Float16)p[nt][r];

    Frag16 pf;
    pf.u4[0] = *(const uint4*)(&Ps[w][ln][g * 8]);
    pf.u4[1] = *(const uint4*)(&Ps[w][ln][g * 8 + 16]);

    // O += P * V : 8 tiles over D=128, K-dim = 32 keys
#pragma unroll
    for (int nd = 0; nd < 8; nd++) {
      Frag16 vf;
      int dcol = nd * 16 + ln;
      vf.u4[0] = *(const uint4*)(&Vs[dcol][g * 8]);
      vf.u4[1] = *(const uint4*)(&Vs[dcol][g * 8 + 16]);
      o[nd] = wmma_f16(pf.v, vf.v, o[nd]);
    }
  }

  // Normalize and write token-major [S, NQ*D] f16
#pragma unroll
  for (int nd = 0; nd < 8; nd++) {
    int d = nd * 16 + ln;
#pragma unroll
    for (int r = 0; r < 8; r++) {
      int m = qrow0 + r + g * 8;
      Out[(long)m * (NQc * Dc) + h * Dc + d] = (_Float16)(o[nd][r] / lrun[r]);
    }
  }
}

// ---------------------------------------------------------------------------
// Host orchestration
// ---------------------------------------------------------------------------
extern "C" void kernel_launch(void* const* d_in, const int* in_sizes, int n_in,
                              void* d_out, int out_size, void* d_ws, size_t ws_size,
                              hipStream_t stream) {
  (void)in_sizes; (void)n_in; (void)out_size; (void)ws_size;

  const float* hidden = (const float*)d_in[0];
  const float* cosp   = (const float*)d_in[1];
  const float* sinp   = (const float*)d_in[2];
  // d_in[3]: attention_mask — causal, computed analytically in-kernel
  const float* Wq  = (const float*)d_in[4];
  const float* bq  = (const float*)d_in[5];
  const float* Wk  = (const float*)d_in[6];
  const float* bk  = (const float*)d_in[7];
  const float* Wv  = (const float*)d_in[8];
  const float* bv  = (const float*)d_in[9];
  const float* Wo  = (const float*)d_in[10];
  const float* Wg  = (const float*)d_in[11];
  const float* Wu  = (const float*)d_in[12];
  const float* Wd  = (const float*)d_in[13];
  const float* ln1 = (const float*)d_in[14];
  const float* ln2 = (const float*)d_in[15];
  const float* lnf = (const float*)d_in[16];

  char* ws = (char*)d_ws;
  size_t off = 0;
  auto alloc = [&](size_t bytes) -> char* {
    char* p = ws + off;
    off += (bytes + 255) & ~(size_t)255;
    return p;
  };

  float*    x     = (float*)   alloc((size_t)Sc * Hc * 4);
  _Float16* hnorm = (_Float16*)alloc((size_t)Sc * Hc * 2);
  _Float16* wf16  = (_Float16*)alloc((size_t)Hc * Ic * 2);  // largest weight
  float*    qb    = (float*)   alloc((size_t)Sc * NQc  * Dc * 4);
  float*    kb    = (float*)   alloc((size_t)Sc * NKVc * Dc * 4);
  float*    vb    = (float*)   alloc((size_t)Sc * NKVc * Dc * 4);
  _Float16* qfb   = (_Float16*)alloc((size_t)NQc  * Sc * Dc * 2);
  _Float16* kfb   = (_Float16*)alloc((size_t)NKVc * Sc * Dc * 2);
  _Float16* vfb   = (_Float16*)alloc((size_t)NKVc * Dc * Sc * 2);
  _Float16* attn  = (_Float16*)alloc((size_t)Sc * NQc * Dc * 2);
  float*    gbuf  = (float*)   alloc((size_t)Sc * Ic * 4);
  float*    ubuf  = (float*)   alloc((size_t)Sc * Ic * 4);
  _Float16* act   = (_Float16*)alloc((size_t)Sc * Ic * 2);

  hipMemcpyAsync(x, hidden, (size_t)Sc * Hc * 4, hipMemcpyDeviceToDevice, stream);

  // Transposing conversion: W[K,N] f32 -> Wt[N,K] f16
  auto convT = [&](const float* src, _Float16* dst, int K, int N) {
    convT_kernel<<<dim3(N / 32, K / 32), dim3(32, 8), 0, stream>>>(src, dst, K, N);
  };

  const int NQD = NQc * Dc;   // 1536
  const int KVD = NKVc * Dc;  // 256

  for (int i = 0; i < Lc; i++) {
    const float* Wqi = Wq + (size_t)i * Hc * NQD;
    const float* Wki = Wk + (size_t)i * Hc * KVD;
    const float* Wvi = Wv + (size_t)i * Hc * KVD;
    const float* Woi = Wo + (size_t)i * NQD * Hc;
    const float* Wgi = Wg + (size_t)i * Hc * Ic;
    const float* Wui = Wu + (size_t)i * Hc * Ic;
    const float* Wdi = Wd + (size_t)i * Ic * Hc;
    const float* bqi = bq + (size_t)i * NQD;
    const float* bki = bk + (size_t)i * KVD;
    const float* bvi = bv + (size_t)i * KVD;

    // --- attention block ---
    rmsnorm_kernel<_Float16><<<Sc, 256, 0, stream>>>(x, ln1 + (size_t)i * Hc, hnorm);

    convT(Wqi, wf16, Hc, NQD);
    gemm_f16_kernel<true, false><<<dim3(NQD / 64, Sc / 128), 256, 0, stream>>>(
        hnorm, wf16, qb, bqi, nullptr, Sc, NQD, Hc);
    convT(Wki, wf16, Hc, KVD);
    gemm_f16_kernel<true, false><<<dim3(KVD / 64, Sc / 128), 256, 0, stream>>>(
        hnorm, wf16, kb, bki, nullptr, Sc, KVD, Hc);
    convT(Wvi, wf16, Hc, KVD);
    gemm_f16_kernel<true, false><<<dim3(KVD / 64, Sc / 128), 256, 0, stream>>>(
        hnorm, wf16, vb, bvi, nullptr, Sc, KVD, Hc);

    rope_kernel<<<(Sc * NQc * 64 + 255) / 256, 256, 0, stream>>>(qb, cosp, sinp, qfb, NQc);
    rope_kernel<<<(Sc * NKVc * 64 + 255) / 256, 256, 0, stream>>>(kb, cosp, sinp, kfb, NKVc);
    v_pack_kernel<<<(NKVc * Dc * Sc + 255) / 256, 256, 0, stream>>>(vb, vfb);

    flash_attn_kernel<<<dim3(Sc / 64, NQc), 128, 0, stream>>>(qfb, kfb, vfb, attn);

    convT(Woi, wf16, NQD, Hc);
    gemm_f16_kernel<false, true><<<dim3(Hc / 64, Sc / 128), 256, 0, stream>>>(
        attn, wf16, x, nullptr, x, Sc, Hc, NQD);  // x = residual + attn @ Wo

    // --- MLP block ---
    rmsnorm_kernel<_Float16><<<Sc, 256, 0, stream>>>(x, ln2 + (size_t)i * Hc, hnorm);

    convT(Wgi, wf16, Hc, Ic);
    gemm_f16_kernel<false, false><<<dim3(Ic / 64, Sc / 128), 256, 0, stream>>>(
        hnorm, wf16, gbuf, nullptr, nullptr, Sc, Ic, Hc);
    convT(Wui, wf16, Hc, Ic);
    gemm_f16_kernel<false, false><<<dim3(Ic / 64, Sc / 128), 256, 0, stream>>>(
        hnorm, wf16, ubuf, nullptr, nullptr, Sc, Ic, Hc);

    swiglu_kernel<<<2048, 256, 0, stream>>>(gbuf, ubuf, act, (long)Sc * Ic);

    convT(Wdi, wf16, Ic, Hc);
    gemm_f16_kernel<false, true><<<dim3(Hc / 64, Sc / 128), 256, 0, stream>>>(
        act, wf16, x, nullptr, x, Sc, Hc, Ic);  // x = residual + mlp
  }

  rmsnorm_kernel<float><<<Sc, 256, 0, stream>>>(x, lnf, (float*)d_out);
}
